// OSD0_Decoder_43301860278696
// MI455X (gfx1250) — 8-bit matrix operand order repaired and checked by reading, compiled, not run on hardware
//
#include <hip/hip_runtime.h>
#include <stdint.h>

#define N_     512
#define RANK_  256
#define BS_    128
#define NW      17
#define STRIDE  17

typedef int v8i __attribute__((ext_vector_type(8)));
typedef int v4i __attribute__((vector_size(16)));
typedef __attribute__((address_space(1))) v4i v4i_g;
typedef __attribute__((address_space(3))) v4i v4i_l;
typedef __attribute__((address_space(3))) void LPtr;

#if defined(__has_builtin)
#if __has_builtin(__builtin_amdgcn_global_load_async_to_lds_b128) && \
    __has_builtin(__builtin_amdgcn_s_wait_asynccnt)
#define HAVE_ASYNC_BUILTIN 1
#endif
#if __has_builtin(__builtin_amdgcn_ballot_w32)
#define BALLOT32(p) __builtin_amdgcn_ballot_w32(p)
#endif
#endif
#ifndef BALLOT32
#define BALLOT32(p) ((uint32_t)__ballot(p))
#endif

__global__ __launch_bounds__(256) void osd0_gf2_kernel(
    const float* __restrict__ llr, const int* __restrict__ pcm,
    const int* __restrict__ syn, float* __restrict__ out)
{
  __shared__ uint32_t Mw[RANK_ * STRIDE];
  union Scratch {
    uint32_t stage[16 * N_];
    struct {
      uint32_t Ub[RANK_ * 4];
      uint32_t Pb[544 * 4];
    } eb;
  };
  __shared__ Scratch u;
  __shared__ float Lv[N_];
  __shared__ int so[N_];
  __shared__ int pvt[16];
  __shared__ int pivAll[RANK_];

  const int t    = threadIdx.x;
  const int b    = blockIdx.x;
  const int lane = t & 31;

  Lv[t]       = llr[(size_t)b * N_ + t];
  Lv[t + 256] = llr[(size_t)b * N_ + t + 256];
  __syncthreads();
  for (int rep = 0; rep < 2; ++rep) {
    const int j = t + rep * 256;
    const float x = Lv[j];
    int cnt = 0;
    for (int k = 0; k < N_; ++k) {
      const float y = Lv[k];
      cnt += (y < x) || (y == x && k < j);
    }
    so[cnt] = j;
  }
  __syncthreads();

  const int* pcmb = pcm + (size_t)b * RANK_ * N_;
  for (int g = 0; g < 16; ++g) {
#if defined(HAVE_ASYNC_BUILTIN)
    for (int p = 0; p < 8; ++p) {
      const int ii = p * 1024 + t * 4;
      __builtin_amdgcn_global_load_async_to_lds_b128(
          (v4i_g*)(pcmb + g * 16 * N_ + ii), (v4i_l*)(&u.stage[ii]), 0, 0);
    }
    __builtin_amdgcn_s_wait_asynccnt(0);
#elif defined(__AMDGCN__)
    for (int p = 0; p < 8; ++p) {
      const int ii = p * 1024 + t * 4;
      const uint32_t lds_off = (uint32_t)(uintptr_t)(LPtr*)(&u.stage[ii]);
      const int* ga = pcmb + g * 16 * N_ + ii;
      asm volatile("global_load_async_to_lds_b128 %0, %1, off"
                   :: "v"(lds_off), "v"(ga) : "memory");
    }
    asm volatile("s_wait_asynccnt 0x0" ::: "memory");
#else
    for (int p = 0; p < 8; ++p) {
      const int ii = p * 1024 + t * 4;
      *(int4*)&u.stage[ii] = *(const int4*)(pcmb + g * 16 * N_ + ii);
    }
#endif
    __syncthreads();
    const int rl = t >> 4, w = t & 15;
    const uint32_t* srow = &u.stage[rl * N_];
    uint32_t wv = 0;
    #pragma unroll
    for (int i = 0; i < 32; ++i)
      wv |= (srow[so[w * 32 + i]] & 1u) << i;
    const int R = g * 16 + rl;
    Mw[R * STRIDE + w] = wv;
    if (w == 0) Mw[R * STRIDE + 16] = (uint32_t)(syn[R * BS_ + b] & 1);
    __syncthreads();
  }

  for (int blk = 0; blk < 16; ++blk) {
    const int r0 = blk * 16;

    for (int k = 0; k < 16; ++k) {
      const int row = r0 + k;
      int piv = 513;
      #pragma unroll
      for (int w = 0; w < NW; ++w) {
        const uint32_t v = Mw[row * STRIDE + w];
        if (piv == 513 && v != 0u) piv = w * 32 + (__ffs(v) - 1);
      }
      if (piv == 513) piv = 0;
      if (t == 0) { pvt[k] = piv; pivAll[row] = piv; }
      if (t >= r0 && t < r0 + 16 && t != row) {
        if ((Mw[t * STRIDE + (piv >> 5)] >> (piv & 31)) & 1u) {
          #pragma unroll
          for (int w = 0; w < NW; ++w)
            Mw[t * STRIDE + w] ^= Mw[row * STRIDE + w];
        }
      }
      __syncthreads();
    }

    {
      const bool inblk = (t >= r0 && t < r0 + 16);
      uint32_t d[4] = {0, 0, 0, 0};
      #pragma unroll
      for (int k = 0; k < 16; ++k) {
        const int p = pvt[k];
        const uint32_t bit =
            inblk ? 0u : ((Mw[t * STRIDE + (p >> 5)] >> (p & 31)) & 1u);
        d[k >> 2] |= bit << (8 * (k & 3));
      }
      u.eb.Ub[t*4+0] = d[0]; u.eb.Ub[t*4+1] = d[1];
      u.eb.Ub[t*4+2] = d[2]; u.eb.Ub[t*4+3] = d[3];

      for (int rep = 0; rep < 3; ++rep) {
        if (rep == 2 && t >= 32) break;
        const int c = (rep == 0) ? t : (rep == 1 ? t + 256 : 512 + t);
        uint32_t e[4] = {0, 0, 0, 0};
        if (c < 513) {
          const int wd = c >> 5, bp = c & 31;
          #pragma unroll
          for (int k = 0; k < 16; ++k) {
            const uint32_t bit = (Mw[(r0 + k) * STRIDE + wd] >> bp) & 1u;
            e[k >> 2] |= bit << (8 * (k & 3));
          }
        }
        u.eb.Pb[c*4+0] = e[0]; u.eb.Pb[c*4+1] = e[1];
        u.eb.Pb[c*4+2] = e[2]; u.eb.Pb[c*4+3] = e[3];
      }
    }
    __syncthreads();

    {
      const int wvid = t >> 5;
      for (int i = 0; i < 34; ++i) {
        const int unit = wvid * 34 + i;
        const int Mt = unit / 17;
        const int wd = unit % 17;
        const int arow = Mt * 16 + (lane & 15);
        const int asel = (lane < 16) ? 0 : 2;
        v8i A = {0, 0, 0, 0, 0, 0, 0, 0};
        A[0] = (int)u.eb.Ub[arow * 4 + asel];
        A[1] = (int)u.eb.Ub[arow * 4 + asel + 1];

        const int c0 = (wd * 2 + 0) * 16 + (lane & 15);
        const int c1 = (wd * 2 + 1) * 16 + (lane & 15);
        v8i B0 = {0, 0, 0, 0, 0, 0, 0, 0};
        v8i B1 = {0, 0, 0, 0, 0, 0, 0, 0};
        {
          B0[0] = (int)u.eb.Pb[c0 * 4 + asel];
          B0[1] = (int)u.eb.Pb[c0 * 4 + asel + 1];
          B1[0] = (int)u.eb.Pb[c1 * 4 + asel];
          B1[1] = (int)u.eb.Pb[c1 * 4 + asel + 1];
        }
        v8i C = {0, 0, 0, 0, 0, 0, 0, 0};
        const v8i D0 = __builtin_amdgcn_wmma_i32_16x16x64_iu8(
            false, A, false, B0, C, false, false);
        const v8i D1 = __builtin_amdgcn_wmma_i32_16x16x64_iu8(
            false, A, false, B1, C, false, false);

        uint32_t bal0[8], bal1[8];
        #pragma unroll
        for (int q = 0; q < 8; ++q)
          bal0[q] = BALLOT32((D0[q] & 1) != 0);
        #pragma unroll
        for (int q = 0; q < 8; ++q)
          bal1[q] = BALLOT32((D1[q] & 1) != 0);

        if (lane == 0) {
          #pragma unroll
          for (int q = 0; q < 8; ++q) {
            const uint32_t mlow = (bal0[q] & 0xFFFFu) | ((bal1[q] & 0xFFFFu) << 16);
            const uint32_t mhi  = (bal0[q] >> 16)     | (bal1[q] & 0xFFFF0000u);
            Mw[(Mt * 16 + q    ) * STRIDE + wd] ^= mlow;
            Mw[(Mt * 16 + 8 + q) * STRIDE + wd] ^= mhi;
          }
        }
      }
    }
    __syncthreads();
  }

  float* ob = out + (size_t)b * N_;
  ob[t] = 0.0f;
  ob[t + 256] = 0.0f;
  __syncthreads();
  const int p = pivAll[t];
  if ((Mw[t * STRIDE + 16] & 1u) && p < N_) ob[so[p]] = 1.0f;
}

extern "C" void kernel_launch(void* const* d_in, const int* in_sizes, int n_in,
                              void* d_out, int out_size, void* d_ws, size_t ws_size,
                              hipStream_t stream) {
  (void)in_sizes; (void)n_in; (void)d_ws; (void)ws_size; (void)out_size;
  const float* llr = (const float*)d_in[0];
  const int*   pcm = (const int*)d_in[1];
  const int*   syn = (const int*)d_in[2];
  float*       out = (float*)d_out;
  osd0_gf2_kernel<<<BS_, 256, 0, stream>>>(llr, pcm, syn, out);
}
